// DeformableMultiHeadedAttention_68169720922400
// MI455X (gfx1250) — matrix operand order repaired and checked by reading, not compiled, not run on hardware
//
#include <hip/hip_runtime.h>
#include <hip/hip_bf16.h>

typedef __attribute__((ext_vector_type(16))) __bf16 v16bf;
typedef __attribute__((ext_vector_type(8)))  __bf16 v8bf;
typedef __attribute__((ext_vector_type(8)))  float  v8f;

#define B_  8
#define M_  2048
#define D_  512
#define H_  8
#define HD_ 64
#define NK_ 8

// W stripe staged per K-half: [64 cols][136 k] bf16. 136*2 = 272 B row stride:
//  multiple of 16 B (b128-aligned reads), 272/4 = 68 banks == 4 (mod 64)
//  -> 16 lanes reading 16 consecutive cols are bank-conflict-free.
#define KH_        128
#define WT_STRIDE  136
// A tile staged by TDM: 128 rows x 32 f32, padded to 36-float row stride
// (TDM pad: 4 DWORDs after every 32 DWORDs) -> conflict-free float4 reads.
#define AT_STRIDE  36

// ---- Tensor Data Mover availability (probe-verified builtins) -------------
#if defined(__has_builtin)
# if __has_builtin(__builtin_amdgcn_tensor_load_to_lds) && \
     __has_builtin(__builtin_amdgcn_s_wait_tensorcnt)
#  define HAVE_TDM 1
# endif
#endif
#ifndef HAVE_TDM
# define HAVE_TDM 0
#endif

#if HAVE_TDM
typedef __attribute__((ext_vector_type(4))) unsigned int u32x4_t;
typedef __attribute__((ext_vector_type(4))) int          i32x4_t;
typedef __attribute__((ext_vector_type(8))) int          i32x8_t;

// Issue a TDM 2D tile load: 32x128 f32 tile from global (row stride = row_stride
// elements) into LDS at lds_off, with 4-DW padding after each 32-DW row.
// D# packed per CDNA5 ISA 8.3/8.4 (group0: count/lds/global/type; group1: dims).
__device__ __forceinline__
void tdm_load_tile_f32(const float* gsrc, unsigned lds_off,
                       int tensor_w, int tensor_h, int row_stride) {
    unsigned long long ga = (unsigned long long)(uintptr_t)gsrc;
    u32x4_t g0;
    g0.x = 1u;                                           // count=1, user mode
    g0.y = lds_off;                                      // lds_addr
    g0.z = (unsigned)(ga & 0xffffffffu);                 // global_addr[31:0]
    g0.w = (unsigned)((ga >> 32) & 0x01ffffffu) | (2u << 30); // [56:32] | type=2
    i32x8_t g1;
    g1[0] = (2 << 16) | (1 << 20) | (4 << 22) | (3 << 25);
            // wg_mask=0 | data_size=4B | pad_enable | interval=32DW | amount=4DW
    g1[1] = (tensor_w & 0xffff) << 16;                   // tensor_dim0[15:0]
    g1[2] = ((tensor_w >> 16) & 0xffff) | ((tensor_h & 0xffff) << 16);
    g1[3] = ((tensor_h >> 16) & 0xffff) | (32 << 16);    // tile_dim0 = 32
    g1[4] = 128;                                         // tile_dim1=128, tile_dim2=0
    g1[5] = row_stride;                                  // tensor_dim0_stride[31:0]
    g1[6] = 0;
    g1[7] = 0;
    i32x4_t z4 = (i32x4_t)0;
#if __clang_major__ >= 23
    i32x8_t z8 = (i32x8_t)0;
    __builtin_amdgcn_tensor_load_to_lds(g0, g1, z4, z4, z8, 0);
#else
    __builtin_amdgcn_tensor_load_to_lds(g0, g1, z4, z4, 0);
#endif
}
#endif // HAVE_TDM

// ---------------------------------------------------------------------------
// GEMM: Y[Rows x Cols] = X[Rows x K] @ W[K x Cols] + bias[Cols]
// bf16 WMMA, f32 accumulate. Block = 256 thr = 8 waves -> 128x64 C-tile.
// Wave: 16 rows x 64 cols = 4 accumulators, 4 v_wmma per K-step, A reused 4x.
// B fragments from LDS-transposed W; A tile TDM-DMA'd into LDS, double-buffered.
// Requires Rows%128==0, K%128==0, Cols%64==0 (true for all calls here).
// ---------------------------------------------------------------------------
__global__ __launch_bounds__(256)
void gemm_bf16_wmma(const float* __restrict__ X, const float* __restrict__ W,
                    const float* __restrict__ bias, float* __restrict__ Y,
                    int Rows, int K, int Cols) {
    __shared__ __align__(16) __bf16 wt[64 * WT_STRIDE];          // ~17 KB
#if HAVE_TDM
    __shared__ __align__(16) float  at[2][128 * AT_STRIDE];      // 36 KB
#endif

    const int lane    = threadIdx.x & 31;
    const int wv      = threadIdx.x >> 5;       // wave 0..7
    const int half    = lane >> 4;              // 0: lanes 0-15, 1: lanes 16-31
    const int ln15    = lane & 15;
    const int colBase = blockIdx.y * 64;
    const int rowTile0= blockIdx.x * 128;
    const int rowTile = rowTile0 + wv * 16;
    const int mrow    = rowTile + ln15;

    v8f acc[4] = {};

#if HAVE_TDM
    int buf = 0;
    if (wv == 0)   // DMA first A tile (rows rowTile0..+128, k 0..32)
        tdm_load_tile_f32(X + (size_t)rowTile0 * K,
                          (unsigned)(uintptr_t)&at[0][0], K, 128, K);
#endif

    for (int k0 = 0; k0 < K; k0 += KH_) {
        // ---- cooperative transpose-stage of W[k0:k0+128, colBase:+64] ----
        for (int idx = threadIdx.x; idx < KH_ * 16; idx += 256) {
            int cg = idx & 15;                  // column group (4 cols)
            int kk = idx >> 4;                  // k within half
            float4 w4 = *(const float4*)(W + (size_t)(k0 + kk) * Cols + colBase + cg * 4);
            wt[(cg * 4 + 0) * WT_STRIDE + kk] = (__bf16)w4.x;
            wt[(cg * 4 + 1) * WT_STRIDE + kk] = (__bf16)w4.y;
            wt[(cg * 4 + 2) * WT_STRIDE + kk] = (__bf16)w4.z;
            wt[(cg * 4 + 3) * WT_STRIDE + kk] = (__bf16)w4.w;
        }
#if HAVE_TDM
        if (k0 == 0 && wv == 0) __builtin_amdgcn_s_wait_tensorcnt(0);
#endif
        __syncthreads();

        for (int kb = k0; kb < k0 + KH_; kb += 32) {
            const bool more = (kb + 32 < K);    // uniform across block
#if HAVE_TDM
            // prefetch next A tile via TDM while computing this one
            if (more && wv == 0)
                tdm_load_tile_f32(X + (size_t)rowTile0 * K + (kb + 32),
                                  (unsigned)(uintptr_t)&at[buf ^ 1][0], K, 128, K);
            const float* ap = &at[buf][(wv * 16 + ln15) * AT_STRIDE + half * 8];
            float4 a0 = ((const float4*)ap)[0];
            float4 a1 = ((const float4*)ap)[1];
            float4 a2 = ((const float4*)(ap + 16))[0];
            float4 a3 = ((const float4*)(ap + 16))[1];
#else
            const float* xa = X + (size_t)mrow * K + kb + half * 8;
            if (more) __builtin_prefetch(xa + 32, 0, 0);
            float4 a0 = ((const float4*)xa)[0];
            float4 a1 = ((const float4*)xa)[1];
            float4 a2 = ((const float4*)(xa + 16))[0];
            float4 a3 = ((const float4*)(xa + 16))[1];
#endif
            // A fragment (16x32 bf16, ISA 7.12.2):
            // half==0: K=kb+{0..7,16..23}; half==1: K=kb+{8..15,24..31}
            float af[16] = { a0.x, a0.y, a0.z, a0.w,  a1.x, a1.y, a1.z, a1.w,
                             a2.x, a2.y, a2.z, a2.w,  a3.x, a3.y, a3.z, a3.w };
            v16bf a;
            #pragma unroll
            for (int i = 0; i < 16; ++i) a[i] = (__bf16)af[i];

            // 4 B fragments from LDS, 4 WMMAs reusing A
            #pragma unroll
            for (int c = 0; c < 4; ++c) {
                const __bf16* bp = &wt[(c * 16 + ln15) * WT_STRIDE + (kb - k0) + half * 8];
                v8bf lo = *(const v8bf*)bp;
                v8bf hi = *(const v8bf*)(bp + 16);
                v16bf b = __builtin_shufflevector(lo, hi, 0, 1, 2, 3, 4, 5, 6, 7,
                                                  8, 9, 10, 11, 12, 13, 14, 15);
                acc[c] = __builtin_amdgcn_wmma_f32_16x16x32_bf16(
                    false, a, false, b, (short)0, acc[c], false, false);
            }
#if HAVE_TDM
            if (more) {                          // publish next A tile
                if (wv == 0) __builtin_amdgcn_s_wait_tensorcnt(0);
                __syncthreads();
                buf ^= 1;
            }
#endif
        }
        __syncthreads();
    }

    // C/D layout: VGPR r -> row = rowTile + half*8 + r, col = ln15
    const int rbase = rowTile + half * 8;
    #pragma unroll
    for (int c = 0; c < 4; ++c) {
        const int   col = colBase + c * 16 + ln15;
        const float bs  = bias[col];
        #pragma unroll
        for (int r = 0; r < 8; ++r)
            Y[(size_t)(rbase + r) * Cols + col] = acc[c][r] + bs;
    }
}

// ---------------------------------------------------------------------------
// avg_pool1d over L, kernel=5, zero padding
// ---------------------------------------------------------------------------
__global__ __launch_bounds__(256)
void avgpool5_kernel(const float* __restrict__ qp, float* __restrict__ out) {
    size_t e = (size_t)blockIdx.x * blockDim.x + threadIdx.x;
    if (e >= (size_t)B_ * M_ * D_) return;
    int d = (int)(e % D_);
    int l = (int)((e / D_) % M_);
    int b = (int)(e / ((size_t)D_ * M_));
    float acc = 0.f;
    #pragma unroll
    for (int i = -2; i <= 2; ++i) {
        int ll = l + i;
        if (ll >= 0 && ll < M_) acc += qp[((size_t)b * M_ + ll) * D_ + d];
    }
    out[e] = acc * 0.2f;
}

// ---------------------------------------------------------------------------
// sign_lens[b] = sum(mask[b,0,:]) - 1
// ---------------------------------------------------------------------------
__global__ __launch_bounds__(256)
void signlen_kernel(const unsigned char* __restrict__ mask, float* __restrict__ sl) {
    __shared__ int s[256];
    int b = blockIdx.x, t = threadIdx.x;
    int acc = 0;
    for (int i = t; i < M_; i += 256) acc += mask[(size_t)b * M_ + i] ? 1 : 0;
    s[t] = acc;
    __syncthreads();
    for (int st = 128; st > 0; st >>= 1) {
        if (t < st) s[t] += s[t + st];
        __syncthreads();
    }
    if (t == 0) sl[b] = (float)s[0] - 1.0f;
}

// ---------------------------------------------------------------------------
// Deformable sampling + 8-way attention. One wave per (b,h,l).
// Lane owns head-dims d0=lane, d1=lane+32 (HD=64). Cross-lane dot via shfl_xor.
// ---------------------------------------------------------------------------
__global__ __launch_bounds__(256)
void deform_attn_kernel(const float* __restrict__ kp, const float* __restrict__ vp,
                        const float* __restrict__ qpool, const float* __restrict__ offs,
                        const float* __restrict__ sl, float* __restrict__ ctx) {
    const int lane = threadIdx.x & 31;
    const int wid  = blockIdx.x * (blockDim.x >> 5) + (threadIdx.x >> 5);
    const int l  = wid % M_;
    const int bh = wid / M_;
    const int h  = bh % H_;
    const int b  = bh / H_;
    if (b >= B_) return;

    const int d0 = lane, d1 = lane + 32;
    const size_t rowBase = ((size_t)b * M_ + l) * D_ + h * HD_;
    const float q0 = qpool[rowBase + d0] * 0.125f;   // 1/sqrt(HD)
    const float q1 = qpool[rowBase + d1] * 0.125f;
    const float slen = sl[b];
    const float* offp = offs + ((size_t)b * M_ + l) * (H_ * NK_) + h * NK_;

    float scores[NK_], sv0[NK_], sv1[NK_];

    #pragma unroll
    for (int nk = 0; nk < NK_; ++nk) {
        float samp = (float)(nk - NK_ / 2) + offp[nk] + (float)l;
        samp = fmodf(samp, slen);
        if (samp < 0.f) samp += slen;                    // jnp.mod semantics
        float xn = samp / (float)(M_ - 1) * 2.0f - 1.0f;
        float ix = ((xn + 1.0f) * (float)M_ - 1.0f) * 0.5f;
        float x0 = floorf(ix);
        float w1 = ix - x0;
        float w0 = 1.0f - w1;
        int x0i = (int)x0;
        int x1i = x0i + 1;
        bool v0ok = (x0i >= 0) && (x0i < M_);
        bool v1ok = (x1i >= 0) && (x1i < M_);
        int c0 = min(max(x0i, 0), M_ - 1);
        int c1 = min(max(x1i, 0), M_ - 1);
        size_t p0 = ((size_t)b * M_ + c0) * D_ + h * HD_;
        size_t p1 = ((size_t)b * M_ + c1) * D_ + h * HD_;

        float k00 = v0ok ? kp[p0 + d0] : 0.f;
        float k01 = v0ok ? kp[p0 + d1] : 0.f;
        float k10 = v1ok ? kp[p1 + d0] : 0.f;
        float k11 = v1ok ? kp[p1 + d1] : 0.f;
        float g00 = v0ok ? vp[p0 + d0] : 0.f;
        float g01 = v0ok ? vp[p0 + d1] : 0.f;
        float g10 = v1ok ? vp[p1 + d0] : 0.f;
        float g11 = v1ok ? vp[p1 + d1] : 0.f;

        float sk0 = 0.5f * (w0 * k00 + w1 * k10);
        float sk1 = 0.5f * (w0 * k01 + w1 * k11);
        sv0[nk]   = 0.5f * (w0 * g00 + w1 * g10);
        sv1[nk]   = 0.5f * (w0 * g01 + w1 * g11);

        float part = q0 * sk0 + q1 * sk1;
        #pragma unroll
        for (int s = 16; s > 0; s >>= 1) part += __shfl_xor(part, s, 32);
        scores[nk] = part;                               // broadcast to all lanes
    }

    float mx = scores[0];
    #pragma unroll
    for (int nk = 1; nk < NK_; ++nk) mx = fmaxf(mx, scores[nk]);
    float den = 0.f;
    #pragma unroll
    for (int nk = 0; nk < NK_; ++nk) { scores[nk] = __expf(scores[nk] - mx); den += scores[nk]; }
    float inv = 1.0f / den;

    float c0acc = 0.f, c1acc = 0.f;
    #pragma unroll
    for (int nk = 0; nk < NK_; ++nk) {
        float w = scores[nk] * inv;
        c0acc += w * sv0[nk];
        c1acc += w * sv1[nk];
    }
    ctx[rowBase + d0] = c0acc;
    ctx[rowBase + d1] = c1acc;
}

// ---------------------------------------------------------------------------
extern "C" void kernel_launch(void* const* d_in, const int* in_sizes, int n_in,
                              void* d_out, int out_size, void* d_ws, size_t ws_size,
                              hipStream_t stream) {
    const float*         k    = (const float*)d_in[0];
    const float*         v    = (const float*)d_in[1];
    const float*         q    = (const float*)d_in[2];
    const unsigned char* mask = (const unsigned char*)d_in[3];
    const float* Wk   = (const float*)d_in[4];
    const float* bk   = (const float*)d_in[5];
    const float* Wv   = (const float*)d_in[6];
    const float* bv   = (const float*)d_in[7];
    const float* Wq   = (const float*)d_in[8];
    const float* bq   = (const float*)d_in[9];
    const float* Woff = (const float*)d_in[10];
    const float* boff = (const float*)d_in[11];
    const float* Wo   = (const float*)d_in[12];
    const float* bo   = (const float*)d_in[13];
    float* out = (float*)d_out;

    const int NROW = B_ * M_;                    // 16384 rows
    float* kp    = (float*)d_ws;
    float* vp    = kp    + (size_t)NROW * D_;
    float* qp    = vp    + (size_t)NROW * D_;
    float* qpool = qp    + (size_t)NROW * D_;
    float* offsw = qpool + (size_t)NROW * D_;    // NROW * (H*NK) = NROW*64
    float* ctx   = offsw + (size_t)NROW * (H_ * NK_);
    float* sl    = ctx   + (size_t)NROW * D_;    // B floats

    const dim3 g512(NROW / 128, D_ / 64);        // 128 x 8 blocks
    const dim3 g64 (NROW / 128, (H_ * NK_) / 64);// 128 x 1 blocks

    // 1-3: input projections
    gemm_bf16_wmma<<<g512, 256, 0, stream>>>(k, Wk, bk, kp, NROW, D_, D_);
    gemm_bf16_wmma<<<g512, 256, 0, stream>>>(v, Wv, bv, vp, NROW, D_, D_);
    gemm_bf16_wmma<<<g512, 256, 0, stream>>>(q, Wq, bq, qp, NROW, D_, D_);

    // 4: avg pool on qp
    size_t nelem = (size_t)NROW * D_;
    avgpool5_kernel<<<(int)((nelem + 255) / 256), 256, 0, stream>>>(qp, qpool);

    // 5: offsets GEMM
    gemm_bf16_wmma<<<g64, 256, 0, stream>>>(qpool, Woff, boff, offsw, NROW, D_, H_ * NK_);

    // 6: sign lengths from mask
    signlen_kernel<<<B_, 256, 0, stream>>>(mask, sl);

    // 7: deformable sampling + attention
    deform_attn_kernel<<<(B_ * H_ * M_) / 8, 256, 0, stream>>>(kp, vp, qpool, offsw, sl, ctx);

    // 8: output projection
    gemm_bf16_wmma<<<g512, 256, 0, stream>>>(ctx, Wo, bo, out, NROW, D_, D_);
}
